// PointNetFeaturePropagation_24120536335104
// MI455X (gfx1250) — matrix operand order repaired and checked by reading, not compiled, not run on hardware
//
#include <hip/hip_runtime.h>

// PointNet Feature Propagation for MI455X (gfx1250, wave32, WMMA).
// Bandwidth-bound op (~17 GFLOP vs ~120 MB traffic @ 23.3 TB/s), so:
//  - 3-NN search: VALU + LDS-staged source points (S=2048 -> 32KB LDS).
//  - MLP: bf16 WMMA (v_wmma_f32_16x16x32_bf16) with f32 accumulation,
//    LayerNorm stats in f32 via half-wave shuffles (each wave owns 16 rows).

#define BN   4
#define NN   16384
#define SS   2048
#define DD1  128
#define DD2  256
#define CIN  384      // DD1 + DD2
#define HD   256      // hidden dim (layer 1 out)
#define OD   128      // output dim (layer 2 out)

typedef __attribute__((ext_vector_type(16))) __bf16       v16bf;
typedef __attribute__((ext_vector_type(8)))  float        v8f;
typedef __attribute__((ext_vector_type(4)))  unsigned int u32x4;

union FragU { v16bf v; u32x4 q[2]; };

__device__ __forceinline__ unsigned short f2bf(float f) {
  unsigned int u = __float_as_uint(f);
  u += 0x7FFFu + ((u >> 16) & 1u);   // round-to-nearest-even
  return (unsigned short)(u >> 16);
}

// A-fragment (16x32 bf16): per lane two 16B chunks at +0 and +16 elements.
__device__ __forceinline__ v16bf load_frag_split(const unsigned short* p) {
  FragU u;
  u.q[0] = *(const u32x4*)(p);
  u.q[1] = *(const u32x4*)(p + 16);
  return u.v;
}
// B-fragment (32x16 bf16): per lane 16 contiguous elements (two 16B chunks).
__device__ __forceinline__ v16bf load_frag_cont(const unsigned short* p) {
  FragU u;
  u.q[0] = *(const u32x4*)(p);
  u.q[1] = *(const u32x4*)(p + 8);
  return u.v;
}

// ---------------------------------------------------------------------------
// Kernel 1: convert weights f32 -> bf16 (stays resident in 192MB L2).
// ---------------------------------------------------------------------------
__global__ void convert_weights_kernel(const float* __restrict__ W1,
                                       const float* __restrict__ W2,
                                       unsigned short* __restrict__ w1bf,
                                       unsigned short* __restrict__ w2bf) {
  int i = blockIdx.x * blockDim.x + threadIdx.x;
  if (i < HD * CIN) w1bf[i] = f2bf(W1[i]);
  if (i < OD * HD)  w2bf[i] = f2bf(W2[i]);
}

// ---------------------------------------------------------------------------
// Kernel 2: 3-NN + inverse-distance interpolation + concat -> X bf16 [B*N,384]
// One thread per query point; all S=2048 source points staged in LDS (32KB).
// ---------------------------------------------------------------------------
__global__ void __launch_bounds__(256)
three_nn_interp_kernel(const float* __restrict__ xyz1,
                       const float* __restrict__ xyz2,
                       const float* __restrict__ points1,
                       const float* __restrict__ points2,
                       unsigned short* __restrict__ xbf) {
  __shared__ float sx[SS], sy[SS], sz[SS], sn[SS];
  const int tid = threadIdx.x;
  const int b   = blockIdx.x >> 6;                  // 64 blocks per batch
  const int n   = ((blockIdx.x & 63) << 8) + tid;

  const float* x2b = xyz2 + (size_t)b * 3 * SS;
  for (int i = tid; i < SS; i += 256) {
    float xx = x2b[i], yy = x2b[SS + i], zz = x2b[2 * SS + i];
    sx[i] = xx; sy[i] = yy; sz[i] = zz;
    sn[i] = xx * xx + yy * yy + zz * zz;
  }
  __syncthreads();

  const float* x1b = xyz1 + (size_t)b * 3 * NN;
  float px = x1b[n], py = x1b[NN + n], pz = x1b[2 * NN + n];
  float pn = px * px + py * py + pz * pz;

  float d0 = 3.4e38f, d1v = 3.4e38f, d2v = 3.4e38f;
  int   i0 = 0, i1 = 0, i2 = 0;
#pragma unroll 4
  for (int s = 0; s < SS; ++s) {
    float dot = px * sx[s] + py * sy[s] + pz * sz[s];
    float d   = pn + sn[s] - 2.0f * dot;
    if (d < d0)       { d2v = d1v; i2 = i1; d1v = d0; i1 = i0; d0 = d; i0 = s; }
    else if (d < d1v) { d2v = d1v; i2 = i1; d1v = d;  i1 = s; }
    else if (d < d2v) { d2v = d;   i2 = s; }
  }
  float w0 = 1.0f / (d0 + 1e-8f);
  float w1 = 1.0f / (d1v + 1e-8f);
  float w2 = 1.0f / (d2v + 1e-8f);
  float ws = 1.0f / (w0 + w1 + w2);
  w0 *= ws; w1 *= ws; w2 *= ws;

  unsigned int* xrow = (unsigned int*)(xbf + (size_t)(b * NN + n) * CIN);

  // cols [0,128): points1 transposed (coalesced across lanes per channel)
  const float* p1b = points1 + (size_t)b * DD1 * NN + n;
#pragma unroll 4
  for (int c = 0; c < DD1; c += 2) {
    unsigned short lo = f2bf(p1b[(size_t)c * NN]);
    unsigned short hi = f2bf(p1b[(size_t)(c + 1) * NN]);
    xrow[c >> 1] = ((unsigned)hi << 16) | lo;
  }
  // cols [128,384): interpolated points2 features (2MB/batch -> L2 resident)
  const float* p2b = points2 + (size_t)b * DD2 * SS;
#pragma unroll 4
  for (int c = 0; c < DD2; c += 2) {
    const float* r0 = p2b + (size_t)c * SS;
    const float* r1 = r0 + SS;
    float v0 = w0 * r0[i0] + w1 * r0[i1] + w2 * r0[i2];
    float v1 = w0 * r1[i0] + w1 * r1[i1] + w2 * r1[i2];
    xrow[(DD1 + c) >> 1] = ((unsigned)f2bf(v1) << 16) | f2bf(v0);
  }
}

// ---------------------------------------------------------------------------
// Kernel 3: fused  X@W1^T + b1 -> LN -> ReLU -> @W2^T + b2 -> LN -> ReLU
// 128 rows per block, 8 waves, each wave owns a 16-row strip end-to-end.
// ---------------------------------------------------------------------------
#define H1STRIDE 272   // 256 + pad (bf16 elems) to break LDS bank conflicts
#define STSTRIDE 132   // 128 + pad (f32)

__global__ void __launch_bounds__(256)
fused_mlp_kernel(const unsigned short* __restrict__ xbf,
                 const unsigned short* __restrict__ w1bf,
                 const unsigned short* __restrict__ w2bf,
                 const float* __restrict__ b1, const float* __restrict__ g1,
                 const float* __restrict__ be1,
                 const float* __restrict__ b2, const float* __restrict__ g2,
                 const float* __restrict__ be2,
                 float* __restrict__ out) {
  __shared__ __align__(16) unsigned char smem[128 * H1STRIDE * 2]; // 69632 B
  unsigned short* h1    = (unsigned short*)smem;  // [128][H1STRIDE] bf16
  float*          stage = (float*)smem;           // [128][STSTRIDE] f32 (reuse)

  const int tid  = threadIdx.x;
  const int wv   = tid >> 5;        // wave id 0..7 (wave32)
  const int lane = tid & 31;
  const int lc   = lane & 15;
  const int hi   = lane >> 4;       // half-wave select
  const int R0   = blockIdx.x * 128;

  v8f vzero = {0.f, 0.f, 0.f, 0.f, 0.f, 0.f, 0.f, 0.f};

  // ---------------- Phase 1: H1 = X @ W1^T  (M=16/wave, N=256, K=384) -----
  v8f acc1[16];
#pragma unroll
  for (int t = 0; t < 16; ++t) acc1[t] = vzero;

  const unsigned short* arow = xbf + (size_t)(R0 + wv * 16 + lc) * CIN;
#pragma unroll
  for (int k = 0; k < 12; ++k) {
    v16bf a = load_frag_cont(arow + 32 * k + (hi ? 16 : 0));
#pragma unroll
    for (int t = 0; t < 16; ++t) {
      const unsigned short* bp =
          w1bf + (size_t)(t * 16 + lc) * CIN + 32 * k + (hi ? 16 : 0);
      v16bf bf = load_frag_cont(bp);
      acc1[t] = __builtin_amdgcn_wmma_f32_16x16x32_bf16(
          false, a, false, bf, (short)0, acc1[t], false, false);
    }
  }

  // LayerNorm(256) + ReLU. Row for (vgpr i, half): wv*16 + i + 8*hi.
  {
    float biasv[16], gainv[16], betav[16];
#pragma unroll
    for (int t = 0; t < 16; ++t) {
      int c = t * 16 + lc;
      biasv[t] = b1[c]; gainv[t] = g1[c]; betav[t] = be1[c];
    }
    float ps[8], pq[8];
#pragma unroll
    for (int i = 0; i < 8; ++i) { ps[i] = 0.f; pq[i] = 0.f; }
#pragma unroll
    for (int t = 0; t < 16; ++t)
#pragma unroll
      for (int i = 0; i < 8; ++i) {
        float v = acc1[t][i] + biasv[t];
        acc1[t][i] = v;
        ps[i] += v; pq[i] += v * v;
      }
#pragma unroll
    for (int m = 1; m < 16; m <<= 1)
#pragma unroll
      for (int i = 0; i < 8; ++i) {
        ps[i] += __shfl_xor(ps[i], m, 16);
        pq[i] += __shfl_xor(pq[i], m, 16);
      }
    float mean1[8], rstd1[8];
#pragma unroll
    for (int i = 0; i < 8; ++i) {
      float mu  = ps[i] * (1.0f / 256.0f);
      float var = pq[i] * (1.0f / 256.0f) - mu * mu;
      mean1[i] = mu;
      rstd1[i] = rsqrtf(var + 1e-5f);
    }
#pragma unroll
    for (int t = 0; t < 16; ++t)
#pragma unroll
      for (int i = 0; i < 8; ++i) {
        float v = (acc1[t][i] - mean1[i]) * rstd1[i] * gainv[t] + betav[t];
        v = fmaxf(v, 0.0f);
        int row = wv * 16 + i + 8 * hi;
        h1[row * H1STRIDE + t * 16 + lc] = f2bf(v);
      }
  }
  __syncthreads();

  // ---------------- Phase 2: H2 = H1 @ W2^T  (N=128, K=256) ---------------
  v8f acc2[8];
#pragma unroll
  for (int t = 0; t < 8; ++t) acc2[t] = vzero;

  const unsigned short* arow2 = h1 + (wv * 16 + lc) * H1STRIDE;
#pragma unroll
  for (int k = 0; k < 8; ++k) {
    v16bf a = load_frag_cont(arow2 + 32 * k + (hi ? 16 : 0));
#pragma unroll
    for (int t = 0; t < 8; ++t) {
      const unsigned short* bp =
          w2bf + (size_t)(t * 16 + lc) * HD + 32 * k + (hi ? 16 : 0);
      v16bf bf = load_frag_cont(bp);
      acc2[t] = __builtin_amdgcn_wmma_f32_16x16x32_bf16(
          false, a, false, bf, (short)0, acc2[t], false, false);
    }
  }

  // LayerNorm(128) + ReLU, stats in registers only.
  float biasv2[8], gainv2[8], betav2[8];
#pragma unroll
  for (int t = 0; t < 8; ++t) {
    int c = t * 16 + lc;
    biasv2[t] = b2[c]; gainv2[t] = g2[c]; betav2[t] = be2[c];
  }
  float ps[8], pq[8];
#pragma unroll
  for (int i = 0; i < 8; ++i) { ps[i] = 0.f; pq[i] = 0.f; }
#pragma unroll
  for (int t = 0; t < 8; ++t)
#pragma unroll
    for (int i = 0; i < 8; ++i) {
      float v = acc2[t][i] + biasv2[t];
      acc2[t][i] = v;
      ps[i] += v; pq[i] += v * v;
    }
#pragma unroll
  for (int m = 1; m < 16; m <<= 1)
#pragma unroll
    for (int i = 0; i < 8; ++i) {
      ps[i] += __shfl_xor(ps[i], m, 16);
      pq[i] += __shfl_xor(pq[i], m, 16);
    }
  float mean2[8], rstd2[8];
#pragma unroll
  for (int i = 0; i < 8; ++i) {
    float mu  = ps[i] * (1.0f / 128.0f);
    float var = pq[i] * (1.0f / 128.0f) - mu * mu;
    mean2[i] = mu;
    rstd2[i] = rsqrtf(var + 1e-5f);
  }

  __syncthreads();   // all h1 reads done before reusing smem as stage
#pragma unroll
  for (int t = 0; t < 8; ++t)
#pragma unroll
    for (int i = 0; i < 8; ++i) {
      float v = (acc2[t][i] - mean2[i]) * rstd2[i] * gainv2[t] + betav2[t];
      v = fmaxf(v, 0.0f);
      int row = wv * 16 + i + 8 * hi;
      stage[row * STSTRIDE + t * 16 + lc] = v;
    }
  __syncthreads();

  // Transposed, coalesced write-out: out[b, c, n] with lanes along n.
  const int b     = R0 / NN;     // 128 | N, blocks never span batches
  const int nbase = R0 % NN;
  for (int e = tid; e < 128 * 128; e += 256) {
    int row = e & 127;
    int c   = e >> 7;
    out[((size_t)(b * OD + c)) * NN + nbase + row] = stage[row * STSTRIDE + c];
  }
}

// ---------------------------------------------------------------------------
extern "C" void kernel_launch(void* const* d_in, const int* in_sizes, int n_in,
                              void* d_out, int out_size, void* d_ws,
                              size_t ws_size, hipStream_t stream) {
  (void)in_sizes; (void)n_in; (void)out_size; (void)ws_size;
  const float* xyz1    = (const float*)d_in[0];
  const float* xyz2    = (const float*)d_in[1];
  const float* points1 = (const float*)d_in[2];
  const float* points2 = (const float*)d_in[3];
  const float* W1      = (const float*)d_in[4];
  const float* b1      = (const float*)d_in[5];
  const float* g1      = (const float*)d_in[6];
  const float* be1     = (const float*)d_in[7];
  const float* W2      = (const float*)d_in[8];
  const float* b2      = (const float*)d_in[9];
  const float* g2      = (const float*)d_in[10];
  const float* be2     = (const float*)d_in[11];
  float* out = (float*)d_out;

  char* ws = (char*)d_ws;
  unsigned short* xbf  = (unsigned short*)ws;                 // [B*N][384] bf16
  size_t xbytes = (size_t)BN * NN * CIN * 2;                  // ~50.3 MB
  unsigned short* w1bf = (unsigned short*)(ws + xbytes);      // [256][384] bf16
  unsigned short* w2bf = w1bf + (size_t)HD * CIN;             // [128][256] bf16

  convert_weights_kernel<<<(HD * CIN + 255) / 256, 256, 0, stream>>>(
      W1, W2, w1bf, w2bf);
  three_nn_interp_kernel<<<BN * NN / 256, 256, 0, stream>>>(
      xyz1, xyz2, points1, points2, xbf);
  fused_mlp_kernel<<<BN * NN / 128, 256, 0, stream>>>(
      xbf, w1bf, w2bf, b1, g1, be1, b2, g2, be2, out);
}
